// HybridSpatialReasoningModule_83940840833729
// MI455X (gfx1250) — matrix operand order repaired and checked by reading, not compiled, not run on hardware
//
#include <hip/hip_runtime.h>

// HybridSpatialReasoningModule for MI455X (gfx1250, wave32, WMMA).
// Pipeline: repack W2 -> KNN (LDS-resident) -> geo MLP w/ v_wmma_f32_16x16x32_f16
//           -> segment MLP -> elementwise combine.

#define BN 4
#define NN 4096
#define DD 768
#define KK 8
#define HH 128
#define SS 64
#define HG 64
#define EPSV 1e-5f

typedef __attribute__((ext_vector_type(16))) _Float16 v16h;
typedef __attribute__((ext_vector_type(8)))  _Float16 v8h;
typedef __attribute__((ext_vector_type(8)))  float    v8f;

// ---------------------------------------------------------------------------
// Kernel 0: repack W2 (f32, 64x768 row-major) into f16 WMMA-B lane layout.
// Layout: for n-tile t (16 cols), K-chunk c (32 rows), lane L, element e:
//   K = 32c + (L<16 ? 0 : 16) + e ; Ncol = 16t + (L&15)
//   linear index = ((t*2+c)*32 + L)*16 + e   -> 32B contiguous per lane.
// ---------------------------------------------------------------------------
__global__ void k_repack_w2(const float* __restrict__ W2, _Float16* __restrict__ wB) {
    int u    = blockIdx.x * 256 + threadIdx.x;   // 0 .. 49151
    int e    = u & 15;
    int lane = (u >> 4) & 31;
    int tc   = u >> 9;                           // t*2 + c
    int t    = tc >> 1, c = tc & 1;
    int k    = c * 32 + ((lane < 16) ? 0 : 16) + e;
    int n    = t * 16 + (lane & 15);
    wB[u] = (_Float16)W2[k * DD + n];
}

// ---------------------------------------------------------------------------
// Kernel 1: exact KNN (K+1 smallest d2 incl. self, drop first).
// One block = 256 query points of one batch; all candidate coords in LDS (48KB).
// ---------------------------------------------------------------------------
__global__ void __launch_bounds__(256) k_knn(const float* __restrict__ coords,
                                             int* __restrict__ knn) {
    __shared__ float sc[NN * 3];
    int b  = blockIdx.x >> 4;
    int n0 = (blockIdx.x & 15) * 256;
    for (int i = threadIdx.x; i < NN * 3; i += 256)
        sc[i] = coords[(size_t)b * NN * 3 + i];
    __syncthreads();

    int n = n0 + threadIdx.x;
    float cx = sc[n*3+0], cy = sc[n*3+1], cz = sc[n*3+2];

    float bd[9]; int bi[9];
#pragma unroll
    for (int q = 0; q < 9; ++q) { bd[q] = 3.4e38f; bi[q] = 0; }

    for (int j = 0; j < NN; ++j) {
        float dx = sc[j*3+0] - cx;
        float dy = sc[j*3+1] - cy;
        float dz = sc[j*3+2] - cz;
        float d2 = dx*dx + dy*dy + dz*dz;
        if (d2 < bd[8]) {
            float v = d2; int vi = j;
#pragma unroll
            for (int q = 0; q < 9; ++q) {
                if (v < bd[q]) {
                    float tb = bd[q]; int ti = bi[q];
                    bd[q] = v; bi[q] = vi; v = tb; vi = ti;
                }
            }
        }
    }
    int base = (b * NN + n) * KK;
#pragma unroll
    for (int q = 0; q < KK; ++q) knn[base + q] = bi[q + 1];
}

// ---------------------------------------------------------------------------
// Kernel 2: geometric encoder.
// Per wave: 2 points -> 16 rows (2 x 8 neighbors).
//   geom(16x4) -> h = relu(LN(geom@W1+b1)) (16x64, VALU)
//   enc = LN(h@W2 + b2) via WMMA f32_16x16x32_f16, two passes (stats+normalize)
//   out[b,p,:] = 0.1 * mean_k(enc)   (written as pure overwrite of d_out)
// ---------------------------------------------------------------------------
__global__ void __launch_bounds__(256) k_geo(
    const float* __restrict__ coords, const int* __restrict__ knn,
    const float* __restrict__ W1, const float* __restrict__ b1,
    const float* __restrict__ g1, const float* __restrict__ bt1,
    const _Float16* __restrict__ wB,
    const float* __restrict__ b2, const float* __restrict__ g2,
    const float* __restrict__ bt2,
    float* __restrict__ out) {

    __shared__ float sW1[4 * HG];
    __shared__ float sb1[HG], sg1[HG], sbt1[HG];
    __shared__ _Float16 sh[8 * 16 * HG];   // per-wave 16x64 f16 h tiles

    int tid = threadIdx.x, wave = tid >> 5, lane = tid & 31;
    if (tid < 4 * HG) sW1[tid] = W1[tid];
    if (tid < HG) { sb1[tid] = b1[tid]; sg1[tid] = g1[tid]; sbt1[tid] = bt1[tid]; }
    __syncthreads();

    int pair = blockIdx.x * 8 + wave;      // 0 .. B*N/2-1
    int bb_  = pair >> 11;                 // batch (2048 pairs per batch)
    int p0   = (pair & 2047) * 2;

    _Float16* hrow = &sh[wave * 16 * HG];

    if (lane < 16) {
        int r  = lane;                     // row 0..15
        int p  = p0 + (r >> 3);
        int kk = r & 7;
        int gp = bb_ * NN + p;
        int ni = knn[gp * KK + kk];
        const float* cp = coords + (size_t)gp * 3;
        const float* cn = coords + ((size_t)bb_ * NN + ni) * 3;
        float rx = cn[0]-cp[0], ry = cn[1]-cp[1], rz = cn[2]-cp[2];
        float rd = sqrtf(rx*rx + ry*ry + rz*rz);
        float ge0 = rd, ge1 = rx, ge2 = ry, ge3 = fminf(rd, 1.0f);

        float s = 0.f, sq = 0.f;
        for (int j = 0; j < HG; ++j) {
            float v = fmaf(ge0, sW1[j],
                      fmaf(ge1, sW1[HG+j],
                      fmaf(ge2, sW1[2*HG+j],
                      fmaf(ge3, sW1[3*HG+j], sb1[j]))));
            s += v; sq += v * v;
        }
        float m    = s * (1.f / HG);
        float var  = sq * (1.f / HG) - m * m;
        float istd = rsqrtf(var + EPSV);
        for (int j = 0; j < HG; ++j) {
            float v = fmaf(ge0, sW1[j],
                      fmaf(ge1, sW1[HG+j],
                      fmaf(ge2, sW1[2*HG+j],
                      fmaf(ge3, sW1[3*HG+j], sb1[j]))));
            float z = (v - m) * istd * sg1[j] + sbt1[j];
            hrow[r * HG + j] = (_Float16)fmaxf(z, 0.f);
        }
    }
    __syncthreads();

    // Load A (16x64 f16) with contiguous k = 16*(lane/16) + e per 32-k chunk (as wB), 2 K-chunks.
    int arow = lane & 15;
    int koff = (lane < 16) ? 0 : 16;
    const _Float16* ap = hrow + arow * HG;
    v16h a0, a1;
    {
        v8h l0 = *reinterpret_cast<const v8h*>(ap + koff);
        v8h l1 = *reinterpret_cast<const v8h*>(ap + koff + 8);
        v8h l2 = *reinterpret_cast<const v8h*>(ap + 32 + koff);
        v8h l3 = *reinterpret_cast<const v8h*>(ap + 32 + koff + 8);
#pragma unroll
        for (int e = 0; e < 8; ++e) {
            a0[e] = l0[e]; a0[e+8] = l1[e];
            a1[e] = l2[e]; a1[e+8] = l3[e];
        }
    }

    int col = lane & 15;

    // ---- Pass 1: per-row sums / sumsq over all 768 columns ----
    float ssum[8], ssq[8];
#pragma unroll
    for (int r = 0; r < 8; ++r) { ssum[r] = 0.f; ssq[r] = 0.f; }

    for (int t = 0; t < 48; ++t) {
        v16h bm0, bm1;
        const v8h* bp0 = reinterpret_cast<const v8h*>(wB + ((size_t)(t*2+0)*32 + lane)*16);
        const v8h* bp1 = reinterpret_cast<const v8h*>(wB + ((size_t)(t*2+1)*32 + lane)*16);
        v8h x0 = bp0[0], x1 = bp0[1], y0 = bp1[0], y1 = bp1[1];
#pragma unroll
        for (int e = 0; e < 8; ++e) {
            bm0[e] = x0[e]; bm0[e+8] = x1[e];
            bm1[e] = y0[e]; bm1[e+8] = y1[e];
        }
        v8f acc = {};
        acc = __builtin_amdgcn_wmma_f32_16x16x32_f16(false, a0, false, bm0,
                                                     (short)0, acc, false, false);
        acc = __builtin_amdgcn_wmma_f32_16x16x32_f16(false, a1, false, bm1,
                                                     (short)0, acc, false, false);
        float bc = b2[t*16 + col];
#pragma unroll
        for (int r = 0; r < 8; ++r) {
            float x = acc[r] + bc;
            ssum[r] += x; ssq[r] += x * x;
        }
    }
    // reduce across the 16 lanes of each half (xor masks stay within halves)
#pragma unroll
    for (int msk = 1; msk < 16; msk <<= 1) {
#pragma unroll
        for (int r = 0; r < 8; ++r) {
            ssum[r] += __shfl_xor(ssum[r], msk, 32);
            ssq[r]  += __shfl_xor(ssq[r],  msk, 32);
        }
    }
    float mean[8], istd[8];
#pragma unroll
    for (int r = 0; r < 8; ++r) {
        float mm = ssum[r] * (1.f / DD);
        float vv = ssq[r] * (1.f / DD) - mm * mm;
        mean[r] = mm; istd[r] = rsqrtf(vv + EPSV);
    }

    // ---- Pass 2: recompute, normalize, mean over K, write 0.1*ctx ----
    int p = p0 + (lane >> 4);
    for (int t = 0; t < 48; ++t) {
        v16h bm0, bm1;
        const v8h* bp0 = reinterpret_cast<const v8h*>(wB + ((size_t)(t*2+0)*32 + lane)*16);
        const v8h* bp1 = reinterpret_cast<const v8h*>(wB + ((size_t)(t*2+1)*32 + lane)*16);
        v8h x0 = bp0[0], x1 = bp0[1], y0 = bp1[0], y1 = bp1[1];
#pragma unroll
        for (int e = 0; e < 8; ++e) {
            bm0[e] = x0[e]; bm0[e+8] = x1[e];
            bm1[e] = y0[e]; bm1[e+8] = y1[e];
        }
        v8f acc = {};
        acc = __builtin_amdgcn_wmma_f32_16x16x32_f16(false, a0, false, bm0,
                                                     (short)0, acc, false, false);
        acc = __builtin_amdgcn_wmma_f32_16x16x32_f16(false, a1, false, bm1,
                                                     (short)0, acc, false, false);
        float bc = b2[t*16 + col];
        float zs = 0.f;
#pragma unroll
        for (int r = 0; r < 8; ++r)
            zs += (acc[r] + bc - mean[r]) * istd[r];
        int cg = t*16 + col;
        float cv = 0.1f * (g2[cg] * (zs * 0.125f) + bt2[cg]);
        out[(size_t)(bb_ * NN + p) * DD + cg] = cv;
    }
}

// ---------------------------------------------------------------------------
// Kernel 3: segment means + MLP + 2x LayerNorm. One block per (b, s).
// Deterministic scan (no atomics -> graph-replay safe).
// ---------------------------------------------------------------------------
__global__ void __launch_bounds__(256) k_seg(
    const float* __restrict__ feat, const int* __restrict__ labels,
    const float* __restrict__ A1, const float* __restrict__ ab1,
    const float* __restrict__ ag1, const float* __restrict__ abt1,
    const float* __restrict__ A2, const float* __restrict__ ab2,
    const float* __restrict__ ag2, const float* __restrict__ abt2,
    float* __restrict__ agg, float* __restrict__ cnt_out) {

    __shared__ float sm[DD];
    __shared__ float sah[HH];
    __shared__ float stmp[256], stmq[256];
    __shared__ float stat[2];

    int b = blockIdx.x >> 6, s = blockIdx.x & 63;
    int t = threadIdx.x;
    const float* fb = feat + (size_t)b * NN * DD;
    const int*   lb = labels + b * NN;

    float r0 = 0.f, r1 = 0.f, r2 = 0.f; int cnt = 0;
    for (int n = 0; n < NN; ++n) {
        if (lb[n] == s) {                       // uniform across block
            const float* fr = fb + (size_t)n * DD;
            r0 += fr[t]; r1 += fr[t + 256]; r2 += fr[t + 512];
            ++cnt;
        }
    }
    float den = fmaxf((float)cnt, 1.0f);
    sm[t] = r0 / den; sm[t + 256] = r1 / den; sm[t + 512] = r2 / den;
    __syncthreads();

    float ahv = 0.f;
    if (t < HH) {
        float a = ab1[t];
        for (int d = 0; d < DD; ++d) a = fmaf(sm[d], A1[d * HH + t], a);
        ahv = a;
    }
    stmp[t] = (t < HH) ? ahv : 0.f;
    stmq[t] = (t < HH) ? ahv * ahv : 0.f;
    __syncthreads();
    if (t == 0) {
        float su = 0.f, sq = 0.f;
        for (int i = 0; i < HH; ++i) { su += stmp[i]; sq += stmq[i]; }
        float m = su / HH, v = sq / HH - m * m;
        stat[0] = m; stat[1] = rsqrtf(v + EPSV);
    }
    __syncthreads();
    if (t < HH) {
        float z = (ahv - stat[0]) * stat[1] * ag1[t] + abt1[t];
        sah[t] = fmaxf(z, 0.f);
    }
    __syncthreads();

    int c0 = t, c1 = t + 256, c2 = t + 512;
    float o0 = ab2[c0], o1 = ab2[c1], o2 = ab2[c2];
    for (int hh = 0; hh < HH; ++hh) {
        float av = sah[hh];
        const float* ar = A2 + (size_t)hh * DD;
        o0 = fmaf(av, ar[c0], o0);
        o1 = fmaf(av, ar[c1], o1);
        o2 = fmaf(av, ar[c2], o2);
    }
    stmp[t] = o0 + o1 + o2;
    stmq[t] = o0*o0 + o1*o1 + o2*o2;
    __syncthreads();
    if (t == 0) {
        float su = 0.f, sq = 0.f;
        for (int i = 0; i < 256; ++i) { su += stmp[i]; sq += stmq[i]; }
        float m = su / DD, v = sq / DD - m * m;
        stat[0] = m; stat[1] = rsqrtf(v + EPSV);
    }
    __syncthreads();

    float* ao = agg + (size_t)(b * SS + s) * DD;
    ao[c0] = (o0 - stat[0]) * stat[1] * ag2[c0] + abt2[c0];
    ao[c1] = (o1 - stat[0]) * stat[1] * ag2[c1] + abt2[c1];
    ao[c2] = (o2 - stat[0]) * stat[1] * ag2[c2] + abt2[c2];
    if (t == 0) cnt_out[b * SS + s] = (float)cnt;
}

// ---------------------------------------------------------------------------
// Kernel 4: enhanced = where(cnt>=2, 0.9*feat + 0.1*agg[label], feat);
//           out += enhanced   (out already holds 0.1*geometric_context)
// ---------------------------------------------------------------------------
__global__ void __launch_bounds__(256) k_combine(
    const float* __restrict__ feat, const int* __restrict__ labels,
    const float* __restrict__ agg, const float* __restrict__ cnt,
    float* __restrict__ out) {
    size_t i   = (size_t)blockIdx.x * 256 + threadIdx.x;   // exact coverage
    size_t row = i / DD;
    int    c   = (int)(i - row * DD);
    int    lbl = labels[row];
    int    b   = (int)(row >> 12);
    float  cp  = cnt[b * SS + lbl];
    float  f   = feat[i];
    float  e   = (cp >= 2.0f)
               ? (0.9f * f + 0.1f * agg[(size_t)(b * SS + lbl) * DD + c])
               : f;
    out[i] = e + out[i];
}

// ---------------------------------------------------------------------------
extern "C" void kernel_launch(void* const* d_in, const int* in_sizes, int n_in,
                              void* d_out, int out_size, void* d_ws, size_t ws_size,
                              hipStream_t stream) {
    const float* coords = (const float*)d_in[0];
    const float* feat   = (const float*)d_in[1];
    const int*   labels = (const int*)d_in[2];
    const float* W1   = (const float*)d_in[3];
    const float* b1   = (const float*)d_in[4];
    const float* g1   = (const float*)d_in[5];
    const float* bt1  = (const float*)d_in[6];
    const float* W2   = (const float*)d_in[7];
    const float* b2   = (const float*)d_in[8];
    const float* g2   = (const float*)d_in[9];
    const float* bt2  = (const float*)d_in[10];
    const float* A1   = (const float*)d_in[11];
    const float* ab1  = (const float*)d_in[12];
    const float* ag1  = (const float*)d_in[13];
    const float* abt1 = (const float*)d_in[14];
    const float* A2   = (const float*)d_in[15];
    const float* ab2  = (const float*)d_in[16];
    const float* ag2  = (const float*)d_in[17];
    const float* abt2 = (const float*)d_in[18];
    float* out = (float*)d_out;

    char* ws = (char*)d_ws;
    int*      knn = (int*)(ws);                               // 512 KB
    _Float16* wB  = (_Float16*)(ws + 524288);                 //  96 KB
    float*    agg = (float*)(ws + 524288 + 98304);            // 768 KB
    float*    cnt = (float*)(ws + 524288 + 98304 + 786432);   //   1 KB

    k_repack_w2<<<192, 256, 0, stream>>>(W2, wB);
    k_knn<<<64, 256, 0, stream>>>(coords, knn);
    k_geo<<<1024, 256, 0, stream>>>(coords, knn, W1, b1, g1, bt1,
                                    wB, b2, g2, bt2, out);
    k_seg<<<256, 256, 0, stream>>>(feat, labels, A1, ab1, ag1, abt1,
                                   A2, ab2, ag2, abt2, agg, cnt);
    k_combine<<<49152, 256, 0, stream>>>(feat, labels, agg, cnt, out);
}
